// TestNet_90280212561934
// MI455X (gfx1250) — matrix operand order repaired and checked by reading, not compiled, not run on hardware
//
#include <hip/hip_runtime.h>

// ---------------------------------------------------------------------------
// CDNA5 (gfx1250) wave32 WMMA types
// ---------------------------------------------------------------------------
typedef __attribute__((ext_vector_type(16))) __bf16 v16bf;
typedef __attribute__((ext_vector_type(8)))  float  v8f;

static __device__ __forceinline__ __bf16 f2bf(float f) {
  // round-to-nearest-even f32 -> bf16
  unsigned u = __builtin_bit_cast(unsigned, f);
  u += 0x7FFFu + ((u >> 16) & 1u);
  unsigned short h = (unsigned short)(u >> 16);
  return __builtin_bit_cast(__bf16, h);
}

// A-fragment lane layout (16-bit A 16x32, ISA 7.12.2):
//   lane half h, element e -> K = 32*kb + h*8 + (e>>3)*16 + (e&7)
// We store all A operands in order K = 32*kb + h*16 + e (same as B) at convert time, so each
// fragment is ONE contiguous 32-byte load:  idx = ((row*kbn + kb)*2 + h)*16 + e
// B-fragment (16-bit B 32x16), order used here: lane half h, element e -> K = 32*kb + h*16 + e
//   -> already a contiguous 32-byte load from a (col-major) K row.

// ---------------------------------------------------------------------------
// Tap-decomposed implicit-GEMM 3x3 conv (pad=1), bf16 in, f32 accum.
//   Xt : ((H+2)*(W+2), Cp) bf16, halo-padded pixel-major, Cp mult of 32
//   Wt : swizzled (9, Cout16, Cp/32, 2, 16) bf16
// One wave computes a 16(out-ch) x 64(pixel) tile; guard-free inner loop.
// ---------------------------------------------------------------------------
__global__ __launch_bounds__(32) void conv3x3_wmma(
    const __bf16* __restrict__ Xt,
    const __bf16* __restrict__ Wt,
    const float*  __restrict__ bias,
    float*        __restrict__ dst,   // (Cout, H, W) f32
    int Cout, int Cout16, int Cp, int H, int W, int do_relu)
{
  const int lane = threadIdx.x;
  const int half = lane >> 4;
  const int l    = lane & 15;
  const int HW   = H * W;
  const int Wp   = W + 2;
  const int n0   = blockIdx.y * 64;          // 64-pixel tile, never crosses a row
  const int py   = n0 / W;
  const int px0  = n0 - py * W;
  const int mrow = blockIdx.x * 16 + l;      // < Cout16, always valid memory
  const int kbn  = Cp >> 5;

  v8f acc[4] = {{}, {}, {}, {}};

  for (int dy = 0; dy < 3; ++dy) {
    for (int dx = 0; dx < 3; ++dx) {
      const int t = dy * 3 + dx;
      // halo coords: real pixel (py+dy-1, px+dx-1) lives at (py+dy, px+dx)+... 
      const __bf16* brow =
          Xt + ((size_t)(py + dy) * Wp + (px0 + dx) + l) * Cp + half * 16;
      const __bf16* ap =
          Wt + (((size_t)(t * Cout16 + mrow) * kbn) * 2 + half) * 16;

      for (int kb = 0; kb < kbn; ++kb) {
        v16bf a = *(const v16bf*)(ap + (size_t)kb * 32);
#pragma unroll
        for (int j = 0; j < 4; ++j) {
          v16bf b = *(const v16bf*)(brow + (size_t)(j * 16) * Cp + kb * 32);
          acc[j] = __builtin_amdgcn_wmma_f32_16x16x32_bf16(
              false, a, false, b, (short)0, acc[j], false, false);
        }
      }
    }
  }

  // C/D layout: acc elem r -> M = r + 8*half, N = lane&15
#pragma unroll
  for (int j = 0; j < 4; ++j) {
    int n = n0 + j * 16 + l;
#pragma unroll
    for (int r = 0; r < 8; ++r) {
      int mo = blockIdx.x * 16 + r + half * 8;
      if (mo < Cout) {
        float v = acc[j][r] + bias[mo];
        if (do_relu) v = v > 0.f ? v : 0.f;
        dst[(size_t)mo * HW + n] = v;
      }
    }
  }
}

// ---------------------------------------------------------------------------
// Dense GEMM D[M,N] = act(A @ B + bias) (+ residual, post-ReLU), bf16 in.
//   A  : (Mp, K) bf16, rows pre-swizzled into A-fragment order (pad rows 0)
//   Bt : (N, K)  bf16 (B transposed), N mult of 64
// One wave computes 16(M) x 64(N); guard-free inner loop, pure 32B loads.
// ---------------------------------------------------------------------------
__global__ __launch_bounds__(32) void gemm_wmma(
    const __bf16* __restrict__ A,
    const __bf16* __restrict__ Bt,
    const float*  __restrict__ bias,
    const float*  __restrict__ resid,
    float*        __restrict__ D,
    int M, int K, int N, int do_relu)
{
  const int lane = threadIdx.x;
  const int half = lane >> 4;
  const int l    = lane & 15;

  const __bf16* ap  = A + (size_t)(blockIdx.x * 16 + l) * K + half * 16;
  const __bf16* bp0 = Bt + (size_t)(blockIdx.y * 64 + l) * K + half * 16;

  v8f acc[4] = {{}, {}, {}, {}};
  for (int kb = 0; kb * 32 < K; ++kb) {
    __builtin_prefetch(ap + (size_t)kb * 32 + 512, 0, 0);
    v16bf a = *(const v16bf*)(ap + (size_t)kb * 32);
#pragma unroll
    for (int j = 0; j < 4; ++j) {
      v16bf b = *(const v16bf*)(bp0 + (size_t)(j * 16) * K + (size_t)kb * 32);
      acc[j] = __builtin_amdgcn_wmma_f32_16x16x32_bf16(
          false, a, false, b, (short)0, acc[j], false, false);
    }
  }

#pragma unroll
  for (int j = 0; j < 4; ++j) {
    int no = blockIdx.y * 64 + j * 16 + l;
#pragma unroll
    for (int r = 0; r < 8; ++r) {
      int mo = blockIdx.x * 16 + r + half * 8;
      if (mo < M) {
        float v = acc[j][r] + bias[no];
        if (do_relu) v = v > 0.f ? v : 0.f;
        if (resid) v += resid[(size_t)mo * N + no];
        D[(size_t)mo * N + no] = v;
      }
    }
  }
}

// ---------------------------------------------------------------------------
// Layout/convert passes (f32 state -> bf16 WMMA operand layouts)
// ---------------------------------------------------------------------------

// (C1,H,W)+(C2,H,W) f32 -> halo-padded ((H+2)*(W+2), Cp) bf16
__global__ __launch_bounds__(256) void cvt_x_halo(
    const float* __restrict__ s1, int C1,
    const float* __restrict__ s2, int C2,
    __bf16* __restrict__ Xt, int Cp, int H, int W)
{
  int Wp = W + 2;
  int total = (H + 2) * Wp * Cp;
  int i = blockIdx.x * blockDim.x + threadIdx.x;
  if (i >= total) return;
  int pixel = i / Cp;
  int c     = i - pixel * Cp;
  int yy = pixel / Wp;
  int xx = pixel - yy * Wp;
  int y = yy - 1, x = xx - 1;
  float v = 0.f;
  if (y >= 0 && y < H && x >= 0 && x < W) {
    int HW = H * W;
    if (c < C1)           v = s1[(size_t)c * HW + (size_t)y * W + x];
    else if (c - C1 < C2) v = s2[(size_t)(c - C1) * HW + (size_t)y * W + x];
  }
  Xt[i] = f2bf(v);
}

// OIHW (Cout,Cin,3,3) f32 -> swizzled (9, Cout16, Cp/32, 2, 16) bf16
__global__ __launch_bounds__(256) void cvt_w(
    const float* __restrict__ w, __bf16* __restrict__ Wt,
    int Cout, int Cin, int Cout16, int Cp)
{
  int total = 9 * Cout16 * Cp;
  int i = blockIdx.x * blockDim.x + threadIdx.x;
  if (i >= total) return;
  int e   = i & 15;
  int h   = (i >> 4) & 1;
  int kbn = Cp >> 5;
  int rm  = i >> 5;              // (t*Cout16 + m)*kbn + kb
  int kb  = rm % kbn;
  int tm  = rm / kbn;            // t*Cout16 + m
  int t   = tm / Cout16;
  int m   = tm - t * Cout16;
  int k   = kb * 32 + h * 16 + e;
  float v = 0.f;
  if (m < Cout && k < Cin) v = w[((size_t)m * Cin + k) * 9 + t];
  Wt[i] = f2bf(v);
}

// (K,N) f32 -> (N,K) bf16 transpose for GEMM B operand
__global__ __launch_bounds__(256) void cvt_wg(
    const float* __restrict__ w, __bf16* __restrict__ Bt, int K, int N)
{
  int i = blockIdx.x * blockDim.x + threadIdx.x;
  int total = N * K;
  if (i >= total) return;
  int n = i / K;
  int k = i - n * K;
  Bt[i] = f2bf(w[(size_t)k * N + n]);
}

// 4-neighbor mean aggregation on the 66x66 grid, emitting the bf16 GEMM A
// operand in swizzled fragment order (Mp=4368 rows, rows >= 4356 zeroed).
__global__ __launch_bounds__(256) void gaggb(
    const float* __restrict__ X, __bf16* __restrict__ out, int C)
{
  int i = blockIdx.x * blockDim.x + threadIdx.x;
  int total = 4368 * C;
  if (i >= total) return;
  int node = i / C;
  int c    = i - node * C;
  // swizzled output position for (node, channel c)
  int kb = c >> 5, r = c & 31;
  int h  = r >> 4;
  int e  = r & 15;
  size_t idx = (((size_t)node * (C >> 5) + kb) * 2 + h) * 16 + e;
  float val = 0.f;
  if (node < 4356) {
    int y = node / 66, x = node - y * 66;
    float s = X[i];
    float cnt = 1.f;
    if (y > 0)  { s += X[i - 66 * C]; cnt += 1.f; }
    if (y < 65) { s += X[i + 66 * C]; cnt += 1.f; }
    if (x > 0)  { s += X[i - C];      cnt += 1.f; }
    if (x < 65) { s += X[i + C];      cnt += 1.f; }
    val = s / cnt;
  }
  out[idx] = f2bf(val);
}

// ---------------------------------------------------------------------------
// f32 helpers (pool / pad / crop / transposed conv)
// ---------------------------------------------------------------------------
__global__ __launch_bounds__(256) void maxpool2(
    const float* __restrict__ in, float* __restrict__ out, int C, int H, int W)
{
  int HO = H >> 1, WO = W >> 1;
  int i = blockIdx.x * blockDim.x + threadIdx.x;
  int total = C * HO * WO;
  if (i >= total) return;
  int c = i / (HO * WO);
  int r = i - c * (HO * WO);
  int y = r / WO;
  int x = r - y * WO;
  const float* p = in + (size_t)c * H * W + (size_t)(2 * y) * W + 2 * x;
  out[i] = fmaxf(fmaxf(p[0], p[1]), fmaxf(p[W], p[W + 1]));
}

// p2:(C,64,64) -> node-major X0:(66*66, C) with zero border (= jnp.pad)
__global__ __launch_bounds__(256) void pad_nodes(
    const float* __restrict__ p2, float* __restrict__ X0, int C)
{
  int i = blockIdx.x * blockDim.x + threadIdx.x;
  int total = 4356 * C;
  if (i >= total) return;
  int node = i / C;
  int c = i - node * C;
  int y = node / 66, x = node - y * 66;
  float v = 0.f;
  if (y >= 1 && y <= 64 && x >= 1 && x <= 64)
    v = p2[(size_t)c * 4096 + (size_t)(y - 1) * 64 + (x - 1)];
  X0[i] = v;
}

// node-major Xn:(66*66,C) -> fe:(C,64,64) cropping the border
__global__ __launch_bounds__(256) void crop_nodes(
    const float* __restrict__ Xn, float* __restrict__ fe, int C)
{
  int i = blockIdx.x * blockDim.x + threadIdx.x;
  int total = C * 4096;
  if (i >= total) return;
  int c = i / 4096;
  int r = i - c * 4096;
  int y = r / 64, x = r - y * 64;
  fe[i] = Xn[(size_t)((y + 1) * 66 + (x + 1)) * C + c];
}

// Transposed conv, 2x2 kernel, stride 2, VALID. w is HWIO: (2,2,Cin,Cout).
__global__ __launch_bounds__(256) void convT2x2(
    const float* __restrict__ in, const float* __restrict__ w,
    const float* __restrict__ bias, float* __restrict__ out,
    int Cin, int Cout, int H, int W)
{
  int HO = 2 * H, WO = 2 * W;
  int i = blockIdx.x * blockDim.x + threadIdx.x;
  int total = Cout * HO * WO;
  if (i >= total) return;
  int co = i / (HO * WO);
  int r  = i - co * (HO * WO);
  int oy = r / WO, ox = r - oy * WO;
  int dy = oy & 1, dx = ox & 1, y = oy >> 1, x = ox >> 1;
  const float* wp = w + (size_t)((dy * 2 + dx) * Cin) * Cout + co;
  const float* ip = in + (size_t)y * W + x;
  float s = bias[co];
  for (int ci = 0; ci < Cin; ++ci)
    s += ip[(size_t)ci * H * W] * wp[(size_t)ci * Cout];
  out[i] = s;
}

// ---------------------------------------------------------------------------
// Orchestration
// ---------------------------------------------------------------------------
extern "C" void kernel_launch(void* const* d_in, const int* in_sizes, int n_in,
                              void* d_out, int out_size, void* d_ws, size_t ws_size,
                              hipStream_t stream) {
  const float* x    = (const float*)d_in[0];
  const float* d1w1 = (const float*)d_in[1];
  const float* d1b1 = (const float*)d_in[2];
  const float* d1w2 = (const float*)d_in[3];
  const float* d1b2 = (const float*)d_in[4];
  const float* d2w1 = (const float*)d_in[5];
  const float* d2b1 = (const float*)d_in[6];
  const float* d2w2 = (const float*)d_in[7];
  const float* d2b2 = (const float*)d_in[8];
  const float* md1w = (const float*)d_in[9];
  const float* md1b = (const float*)d_in[10];
  const float* md2w = (const float*)d_in[11];
  const float* md2b = (const float*)d_in[12];
  const float* mbw  = (const float*)d_in[13];
  const float* mbb  = (const float*)d_in[14];
  const float* mu1w = (const float*)d_in[15];
  const float* mu1b = (const float*)d_in[16];
  const float* mu2w = (const float*)d_in[17];
  const float* mu2b = (const float*)d_in[18];
  const float* u3tw = (const float*)d_in[19];
  const float* u3tb = (const float*)d_in[20];
  const float* u3w1 = (const float*)d_in[21];
  const float* u3b1 = (const float*)d_in[22];
  const float* u3w2 = (const float*)d_in[23];
  const float* u3b2 = (const float*)d_in[24];
  const float* u4tw = (const float*)d_in[25];
  const float* u4tb = (const float*)d_in[26];
  const float* u4w1 = (const float*)d_in[27];
  const float* u4b1 = (const float*)d_in[28];
  const float* u4w2 = (const float*)d_in[29];
  const float* u4b2 = (const float*)d_in[30];
  const float* ow   = (const float*)d_in[31];
  const float* ob   = (const float*)d_in[32];
  (void)in_sizes; (void)n_in; (void)out_size; (void)ws_size;

  // byte bump allocator, 256B aligned
  char*  base = (char*)d_ws;
  size_t off  = 0;
  auto allocB = [&](size_t bytes) {
    bytes = (bytes + 255) & ~(size_t)255;
    char* p = base + off;
    off += bytes;
    return p;
  };
  // f32 state
  float* B1  = (float*)allocB(4194304ull * 4); // 64x256x256 (skip for u4)
  float* T1  = (float*)allocB(4194304ull * 4); // t1 -> later u4
  float* B2  = (float*)allocB(2097152ull * 4); // 128x128x128 (skip for u3)
  float* T2  = (float*)allocB(2097152ull * 4); // t2 -> later t3
  float* U3  = (float*)allocB(2097152ull * 4); // u3 -> later b3
  float* P1F = (float*)allocB(1048576ull * 4); // p1 -> later fe2 (256x64x64)
  float* P2  = (float*)allocB(524288ull  * 4); // 128x64x64
  float* X0  = (float*)allocB(557568ull  * 4); // 4356x128 node features
  float* BP1 = (float*)allocB(1115136ull * 4); // 4356x256
  float* BP2 = (float*)allocB(2230272ull * 4); // 4356x512
  float* XB  = (float*)allocB(4460544ull * 4); // 4356x1024 -> later b4
  float* XU1 = (float*)allocB(2230272ull * 4); // 4356x512
  float* XU2 = (float*)allocB(1115136ull * 4); // 4356x256
  float* T4  = (float*)allocB(4194304ull * 4); // t4 (64x256x256)
  // bf16 WMMA operand staging
  __bf16* XT   = (__bf16*)allocB(8520192ull * 2); // max (258*258, Cp=128)
  __bf16* WT   = (__bf16*)allocB(294912ull  * 2); // max (9,128,256)
  __bf16* AGGB = (__bf16*)allocB(4472832ull * 2); // (4368, 1024)
  __bf16* BTW  = (__bf16*)allocB(524288ull  * 2); // max (1024,512)

  dim3 wblk(32);
  auto nb = [](long long total) { return dim3((unsigned)((total + 255) / 256)); };

  // full conv layer: convert input + weights, run WMMA conv
  auto conv = [&](const float* s1, int C1, const float* s2, int C2,
                  const float* w, const float* b, float* d,
                  int Cout, int H, int W, int relu) {
    int Cin = C1 + C2;
    int Cp  = (Cin + 31) & ~31;
    int Cout16 = (Cout + 15) & ~15;
    cvt_x_halo<<<nb((long long)(H + 2) * (W + 2) * Cp), 256, 0, stream>>>(
        s1, C1, s2, C2, XT, Cp, H, W);
    cvt_w<<<nb((long long)9 * Cout16 * Cp), 256, 0, stream>>>(
        w, WT, Cout, Cin, Cout16, Cp);
    dim3 g(Cout16 / 16, (H * W) / 64);
    conv3x3_wmma<<<g, wblk, 0, stream>>>(XT, WT, b, d, Cout, Cout16, Cp, H, W, relu);
  };
  // full graph-conv layer: aggregate (-> swizzled bf16 A), transpose W, GEMM
  auto gconv = [&](const float* Xin, const float* w, const float* b,
                   const float* resid, float* D, int K, int N) {
    gaggb<<<nb(4368LL * K), 256, 0, stream>>>(Xin, AGGB, K);
    cvt_wg<<<nb((long long)N * K), 256, 0, stream>>>(w, BTW, K, N);
    dim3 g(4368 / 16, N / 64);
    gemm_wmma<<<g, wblk, 0, stream>>>(AGGB, BTW, b, resid, D, 4356, K, N, 1);
  };

  // ---- encoder ----
  conv(x,   3, nullptr, 0, d1w1, d1b1, T1, 64, 256, 256, 1);
  conv(T1, 64, nullptr, 0, d1w2, d1b2, B1, 64, 256, 256, 1);
  maxpool2<<<nb(64LL * 128 * 128), 256, 0, stream>>>(B1, P1F, 64, 256, 256);
  conv(P1F, 64, nullptr, 0, d2w1, d2b1, T2, 128, 128, 128, 1);
  conv(T2, 128, nullptr, 0, d2w2, d2b2, B2, 128, 128, 128, 1);
  maxpool2<<<nb(128LL * 64 * 64), 256, 0, stream>>>(B2, P2, 128, 128, 128);

  // ---- graph bottleneck on 66x66 padded grid ----
  pad_nodes<<<nb(4356LL * 128), 256, 0, stream>>>(P2, X0, 128);
  gconv(X0,  md1w, md1b, nullptr, BP1, 128, 256);
  gconv(BP1, md2w, md2b, nullptr, BP2, 256, 512);
  gconv(BP2, mbw,  mbb,  nullptr, XB,  512, 1024);
  gconv(XB,  mu1w, mu1b, BP2,     XU1, 1024, 512); // + bp2 (post-ReLU)
  gconv(XU1, mu2w, mu2b, BP1,     XU2, 512,  256); // + bp1 (post-ReLU)
  crop_nodes<<<nb(256LL * 4096), 256, 0, stream>>>(XU2, P1F, 256); // fe2

  // ---- decoder ----
  convT2x2<<<nb(128LL * 128 * 128), 256, 0, stream>>>(P1F, u3tw, u3tb, U3, 256, 128, 64, 64);
  conv(B2, 128, U3, 128, u3w1, u3b1, T2, 128, 128, 128, 1);    // concat [b2,u3]
  conv(T2, 128, nullptr, 0, u3w2, u3b2, U3, 128, 128, 128, 1); // b3 reuses U3

  convT2x2<<<nb(64LL * 256 * 256), 256, 0, stream>>>(U3, u4tw, u4tb, T1, 128, 64, 128, 128);
  conv(B1, 64, T1, 64, u4w1, u4b1, T4, 64, 256, 256, 1);       // concat [b1,u4]
  conv(T4, 64, nullptr, 0, u4w2, u4b2, XB, 64, 256, 256, 1);   // b4 in XB

  // ---- output head (Cout=1, no relu) ----
  conv(XB, 64, nullptr, 0, ow, ob, (float*)d_out, 1, 256, 256, 0);
}
